// DRL4TP_46059229282788
// MI455X (gfx1250) — matrix operand order repaired and checked by reading, not compiled, not run on hardware
//
#include <hip/hip_runtime.h>
#include <math.h>

typedef __bf16 bf16;
typedef bf16  v16bf __attribute__((ext_vector_type(16)));
typedef float v8f   __attribute__((ext_vector_type(8)));

constexpr int Bc = 128;
constexpr int Sc = 1000;
constexpr int Hc = 256;
constexpr int MT  = 32;                    // s-rows per GEMM block (2 x 16 sub-tiles)
constexpr int NSTM = (Sc + MT - 1) / MT;   // 32 s-tiles of 32
constexpr int KT  = Hc / 32;               // 8 k-steps
constexpr int WPK_ELEMS = 16 * KT * 32 * 16;  // 65536 bf16 per packed matrix

// ---------- helpers ----------

__device__ __forceinline__ float half_reduce16(float x) {
    // reduce across the 16 lanes of a half-wave (wave32; xor masks stay in-half)
    x += __shfl_xor(x, 1);
    x += __shfl_xor(x, 2);
    x += __shfl_xor(x, 4);
    x += __shfl_xor(x, 8);
    return x;
}

__device__ __forceinline__ float sigmoidf_(float x) { return 1.0f / (1.0f + __expf(-x)); }

// Stage one MT x Hc f32 tile of `stat` into LDS via async global->LDS DMA,
// then convert to bf16 (zero-padding rows past S).
__device__ __forceinline__ void stage_tile_async(const float* __restrict__ stat,
                                                 int b, int s0,
                                                 float (*Af32)[Hc], bf16 (*At)[Hc]) {
    const int tid = threadIdx.x;
    const int col = (tid & 15) * 16;
#pragma unroll
    for (int r2 = 0; r2 < MT / 16; ++r2) {
        const int row = (tid >> 4) + 16 * r2;
        const int s = s0 + row;
        if (s < Sc) {
            const float* g = stat + ((size_t)b * Sc + s) * Hc + col;
            unsigned lds0 = (unsigned)(uintptr_t)(&Af32[row][col]);
#pragma unroll
            for (int c = 0; c < 4; ++c) {
                asm volatile("global_load_async_to_lds_b128 %0, %1, off"
                             :: "v"(lds0 + 16u * c), "v"(g + 4 * c)
                             : "memory");
            }
        }
    }
    asm volatile("s_wait_asynccnt 0x0" ::: "memory");
    __syncthreads();
#pragma unroll
    for (int r2 = 0; r2 < MT / 16; ++r2) {
        const int row = (tid >> 4) + 16 * r2;
        const int s = s0 + row;
#pragma unroll
        for (int i = 0; i < 16; ++i) {
            float v = (s < Sc) ? Af32[row][col + i] : 0.0f;
            At[row][col + i] = (bf16)v;
        }
    }
}

// A-fragment (16x32 bf16, contiguous k order: element s = k0 + 16*lh + s) from a 16-row LDS sub-tile.
__device__ __forceinline__ v16bf load_a_frag(const bf16 (*At)[Hc], int lm, int lh, int k0) {
    v16bf a;
    const bf16* p0 = &At[lm][k0 + 16 * lh];
    const bf16* p1 = &At[lm][k0 + 16 * lh + 8];
#pragma unroll
    for (int i = 0; i < 8; ++i) { a[i] = p0[i]; a[8 + i] = p1[i]; }
    return a;
}

// ---------- kernel 0: pack W[:, :H] into bf16 B-fragment order ----------
// frag index f = ht*KT + kt ; per lane: 16 bf16 = W[ht*16 + (lane&15)][kt*32 + 16*(lane>>4) + i]
__global__ void pack_w_kernel(const float* __restrict__ W, int ldw, bf16* __restrict__ out) {
    const int f    = blockIdx.x;          // 0 .. 16*KT-1
    const int ht   = f >> 3;
    const int kt   = f & 7;
    const int t    = threadIdx.x;
    const int lane = t >> 3;              // 0..31
    const int i0   = (t & 7) * 2;         // 0,2,..,14
    const int n    = ht * 16 + (lane & 15);
    const int kk   = kt * 32 + 16 * (lane >> 4) + i0;
    const float* src = W + (size_t)n * ldw + kk;
    bf16* dst = out + (((size_t)f * 32) + lane) * 16 + i0;
    dst[0] = (bf16)src[0];
    dst[1] = (bf16)src[1];
}

// ---------- kernel 1: GRU step ----------
__global__ void gru_kernel(const float* __restrict__ dec, const float* __restrict__ lhh,
                           const float* __restrict__ Wih, const float* __restrict__ Whh,
                           const float* __restrict__ bih, const float* __restrict__ bhh,
                           float* __restrict__ hnew, float* __restrict__ out_hh) {
    const int b = blockIdx.x, j = threadIdx.x;
    __shared__ float x[Hc], h[Hc];
    x[j] = dec[(size_t)b * Hc + j];
    h[j] = lhh[(size_t)b * Hc + j];
    __syncthreads();
    float gi[3], gh[3];
#pragma unroll
    for (int g = 0; g < 3; ++g) {
        const float* wi = Wih + (size_t)(g * Hc + j) * Hc;
        const float* wh = Whh + (size_t)(g * Hc + j) * Hc;
        float si = bih[g * Hc + j], sh = bhh[g * Hc + j];
        for (int k = 0; k < Hc; ++k) { si = fmaf(wi[k], x[k], si); sh = fmaf(wh[k], h[k], sh); }
        gi[g] = si; gh[g] = sh;
    }
    float r  = sigmoidf_(gi[0] + gh[0]);
    float z  = sigmoidf_(gi[1] + gh[1]);
    float n  = tanhf(gi[2] + r * gh[2]);
    float hn = (1.0f - z) * n + z * h[j];
    hnew[(size_t)b * Hc + j]   = hn;
    out_hh[(size_t)b * Hc + j] = hn;
}

// ---------- kernel 2: per-batch bias GEMVs ----------
__global__ void bias_kernel(const float* __restrict__ hnew, const float* __restrict__ tgt,
                            const float* __restrict__ se,
                            const float* __restrict__ Wenc, const float* __restrict__ Wtgt,
                            const float* __restrict__ Wptr,
                            float* __restrict__ biasE, float* __restrict__ biasT,
                            float* __restrict__ biasS) {
    const int b = blockIdx.x, hcol = threadIdx.x;
    __shared__ float hv[Hc], tv[Hc], sv[Hc];
    hv[hcol] = hnew[(size_t)b * Hc + hcol];
    tv[hcol] = tgt[(size_t)b * Hc + hcol];
    sv[hcol] = se[(size_t)b * Hc + hcol];
    __syncthreads();
    const float* we = Wenc + (size_t)hcol * (2 * Hc) + Hc;
    const float* wt = Wtgt + (size_t)hcol * (2 * Hc) + Hc;
    const float* wp = Wptr + (size_t)hcol * (3 * Hc) + 2 * Hc;
    float bE = 0.f, bT = 0.f, bS = 0.f;
    for (int k = 0; k < Hc; ++k) {
        bE = fmaf(we[k], hv[k], bE);
        bT = fmaf(wt[k], tv[k], bT);
        bS = fmaf(wp[k], sv[k], bS);
    }
    biasE[(size_t)b * Hc + hcol] = bE;
    biasT[(size_t)b * Hc + hcol] = bT;
    biasS[(size_t)b * Hc + hcol] = bS;
}

// ---------- kernel 3: dual attention-score GEMM (WMMA bf16, packed W, 32-row M-block) ----------
__global__ __launch_bounds__(256) void attn_scores_kernel(
        const float* __restrict__ stat,
        const bf16* __restrict__ WpkE, const bf16* __restrict__ WpkT,
        const float* __restrict__ biasE, const float* __restrict__ biasT,
        const float* __restrict__ vE, const float* __restrict__ vT,
        float* __restrict__ scoreE, float* __restrict__ scoreT) {
    const int b  = blockIdx.x / NSTM;
    const int st = blockIdx.x % NSTM;
    const int s0 = st * MT;

    __shared__ float Af32[MT][Hc];   // 32 KB async-staged f32 tile
    __shared__ bf16  At[MT][Hc];     // 16 KB bf16 tile
    __shared__ float sE[MT], sT[MT];

    const int tid = threadIdx.x;
    if (tid < MT) { sE[tid] = 0.f; sT[tid] = 0.f; }
    stage_tile_async(stat, b, s0, Af32, At);
    __syncthreads();

    const int wave = tid >> 5;            // 0..7
    const int lane = tid & 31;
    const int lm   = lane & 15;
    const int lh   = lane >> 4;
    const int n0   = wave * 16;           // N-tile 0
    const int n1   = (wave + 8) * 16;     // N-tile 1
    const int f0   = wave * KT;           // packed fragment bases
    const int f1   = (wave + 8) * KT;

    const v16bf* fE = (const v16bf*)WpkE;
    const v16bf* fT = (const v16bf*)WpkT;

    v8f aE[2][2] = {{{}, {}}, {{}, {}}};  // [m-subtile][n-tile]
    v8f aT[2][2] = {{{}, {}}, {{}, {}}};

#pragma unroll
    for (int kt = 0; kt < KT; ++kt) {
        v16bf a0  = load_a_frag(At,      lm, lh, kt * 32);   // rows 0..15
        v16bf a1  = load_a_frag(At + 16, lm, lh, kt * 32);   // rows 16..31
        v16bf be0 = fE[(size_t)(f0 + kt) * 32 + lane];
        v16bf be1 = fE[(size_t)(f1 + kt) * 32 + lane];
        v16bf bt0 = fT[(size_t)(f0 + kt) * 32 + lane];
        v16bf bt1 = fT[(size_t)(f1 + kt) * 32 + lane];
        aE[0][0] = __builtin_amdgcn_wmma_f32_16x16x32_bf16(false, a0, false, be0, (short)0, aE[0][0], false, false);
        aE[1][0] = __builtin_amdgcn_wmma_f32_16x16x32_bf16(false, a1, false, be0, (short)0, aE[1][0], false, false);
        aE[0][1] = __builtin_amdgcn_wmma_f32_16x16x32_bf16(false, a0, false, be1, (short)0, aE[0][1], false, false);
        aE[1][1] = __builtin_amdgcn_wmma_f32_16x16x32_bf16(false, a1, false, be1, (short)0, aE[1][1], false, false);
        aT[0][0] = __builtin_amdgcn_wmma_f32_16x16x32_bf16(false, a0, false, bt0, (short)0, aT[0][0], false, false);
        aT[1][0] = __builtin_amdgcn_wmma_f32_16x16x32_bf16(false, a1, false, bt0, (short)0, aT[1][0], false, false);
        aT[0][1] = __builtin_amdgcn_wmma_f32_16x16x32_bf16(false, a0, false, bt1, (short)0, aT[0][1], false, false);
        aT[1][1] = __builtin_amdgcn_wmma_f32_16x16x32_bf16(false, a1, false, bt1, (short)0, aT[1][1], false, false);
    }

    // epilogue: bias + tanh + v-dot, reduce over N
    auto epilogue = [&](v8f (&A)[2][2], const float* __restrict__ bias,
                        const float* __restrict__ v, float* __restrict__ sdst) {
        const float bb0 = bias[(size_t)b * Hc + n0 + lm];
        const float bb1 = bias[(size_t)b * Hc + n1 + lm];
        const float vv0 = v[n0 + lm];
        const float vv1 = v[n1 + lm];
#pragma unroll
        for (int ms = 0; ms < 2; ++ms) {
#pragma unroll
            for (int i = 0; i < 8; ++i) {
                float part = vv0 * tanhf(A[ms][0][i] + bb0) + vv1 * tanhf(A[ms][1][i] + bb1);
                part = half_reduce16(part);
                if (lm == 0) atomicAdd(&sdst[ms * 16 + lh * 8 + i], part);  // row = ms*16 + lh*8 + i
            }
        }
    };
    epilogue(aE, biasE, vE, sE);
    epilogue(aT, biasT, vT, sT);
    __syncthreads();

    if (tid < MT) {
        const int s = s0 + tid;
        if (s < Sc) {
            scoreE[(size_t)b * Sc + s] = sE[tid];
            scoreT[(size_t)b * Sc + s] = sT[tid];
        }
    }
}

// ---------- kernel 4: softmax + context + pointer biases ----------
__global__ void softmax_ctx_kernel(const float* __restrict__ stat,
                                   const float* __restrict__ scoreE, const float* __restrict__ scoreT,
                                   const float* __restrict__ Wptr, const float* __restrict__ biasS,
                                   float* __restrict__ biasPE, float* __restrict__ biasPT) {
    const int b = blockIdx.x, tid = threadIdx.x;
    __shared__ float aE[Sc], aT[Sc];
    __shared__ float red[256];
    __shared__ float ctxE[Hc], ctxT[Hc];

    for (int s = tid; s < Sc; s += 256) {
        aE[s] = scoreE[(size_t)b * Sc + s];
        aT[s] = scoreT[(size_t)b * Sc + s];
    }
    __syncthreads();

    float mE = -1e30f, mT = -1e30f;
    for (int s = tid; s < Sc; s += 256) { mE = fmaxf(mE, aE[s]); mT = fmaxf(mT, aT[s]); }
    red[tid] = mE; __syncthreads();
    for (int o = 128; o > 0; o >>= 1) { if (tid < o) red[tid] = fmaxf(red[tid], red[tid + o]); __syncthreads(); }
    mE = red[0]; __syncthreads();
    red[tid] = mT; __syncthreads();
    for (int o = 128; o > 0; o >>= 1) { if (tid < o) red[tid] = fmaxf(red[tid], red[tid + o]); __syncthreads(); }
    mT = red[0]; __syncthreads();

    float seS = 0.f, stS = 0.f;
    for (int s = tid; s < Sc; s += 256) {
        float e1 = __expf(aE[s] - mE); aE[s] = e1; seS += e1;
        float e2 = __expf(aT[s] - mT); aT[s] = e2; stS += e2;
    }
    red[tid] = seS; __syncthreads();
    for (int o = 128; o > 0; o >>= 1) { if (tid < o) red[tid] += red[tid + o]; __syncthreads(); }
    seS = red[0]; __syncthreads();
    red[tid] = stS; __syncthreads();
    for (int o = 128; o > 0; o >>= 1) { if (tid < o) red[tid] += red[tid + o]; __syncthreads(); }
    stS = red[0]; __syncthreads();
    const float invE = 1.0f / seS, invT = 1.0f / stS;

    // context[h] = sum_s attn[s] * static[b,s,h] (coalesced over h = tid)
    float cE = 0.f, cT = 0.f;
    const float* sp = stat + (size_t)b * Sc * Hc + tid;
    for (int s = 0; s < Sc; ++s) {
        float x = sp[(size_t)s * Hc];
        cE = fmaf(aE[s], x, cE);
        cT = fmaf(aT[s], x, cT);
    }
    ctxE[tid] = cE * invE;
    ctxT[tid] = cT * invT;
    __syncthreads();

    // pointer bias: biasP*(b,h) = biasS + W_ptr[h, H:2H] . ctx*
    const float* wrow = Wptr + (size_t)tid * (3 * Hc) + Hc;
    float bE = biasS[(size_t)b * Hc + tid], bT = bE;
    for (int k = 0; k < Hc; ++k) {
        float w = wrow[k];
        bE = fmaf(w, ctxE[k], bE);
        bT = fmaf(w, ctxT[k], bT);
    }
    biasPE[(size_t)b * Hc + tid] = bE;
    biasPT[(size_t)b * Hc + tid] = bT;
}

// ---------- kernel 5: pointer GEMM (WMMA bf16, packed W, 32-row M-block), dual epilogue ----------
__global__ __launch_bounds__(256) void ptr_kernel(
        const float* __restrict__ stat, const bf16* __restrict__ WpkP,
        const float* __restrict__ biasPE, const float* __restrict__ biasPT,
        const float* __restrict__ vP, float* __restrict__ out) {
    const int b  = blockIdx.x / NSTM;
    const int st = blockIdx.x % NSTM;
    const int s0 = st * MT;

    __shared__ float Af32[MT][Hc];
    __shared__ bf16  At[MT][Hc];
    __shared__ float sE[MT], sT[MT];

    const int tid = threadIdx.x;
    if (tid < MT) { sE[tid] = 0.f; sT[tid] = 0.f; }
    stage_tile_async(stat, b, s0, Af32, At);
    __syncthreads();

    const int wave = tid >> 5;
    const int lane = tid & 31;
    const int lm   = lane & 15;
    const int lh   = lane >> 4;
    const int n0   = wave * 16;
    const int n1   = (wave + 8) * 16;
    const int f0   = wave * KT;
    const int f1   = (wave + 8) * KT;

    const v16bf* fP = (const v16bf*)WpkP;

    v8f acc[2][2] = {{{}, {}}, {{}, {}}};  // [m-subtile][n-tile]
#pragma unroll
    for (int kt = 0; kt < KT; ++kt) {
        v16bf a0 = load_a_frag(At,      lm, lh, kt * 32);
        v16bf a1 = load_a_frag(At + 16, lm, lh, kt * 32);
        v16bf b0 = fP[(size_t)(f0 + kt) * 32 + lane];
        v16bf b1 = fP[(size_t)(f1 + kt) * 32 + lane];
        acc[0][0] = __builtin_amdgcn_wmma_f32_16x16x32_bf16(false, a0, false, b0, (short)0, acc[0][0], false, false);
        acc[1][0] = __builtin_amdgcn_wmma_f32_16x16x32_bf16(false, a1, false, b0, (short)0, acc[1][0], false, false);
        acc[0][1] = __builtin_amdgcn_wmma_f32_16x16x32_bf16(false, a0, false, b1, (short)0, acc[0][1], false, false);
        acc[1][1] = __builtin_amdgcn_wmma_f32_16x16x32_bf16(false, a1, false, b1, (short)0, acc[1][1], false, false);
    }

    const float bE0 = biasPE[(size_t)b * Hc + n0 + lm];
    const float bE1 = biasPE[(size_t)b * Hc + n1 + lm];
    const float bT0 = biasPT[(size_t)b * Hc + n0 + lm];
    const float bT1 = biasPT[(size_t)b * Hc + n1 + lm];
    const float vv0 = vP[n0 + lm];
    const float vv1 = vP[n1 + lm];
#pragma unroll
    for (int ms = 0; ms < 2; ++ms) {
#pragma unroll
        for (int i = 0; i < 8; ++i) {
            float pe = vv0 * tanhf(acc[ms][0][i] + bE0) + vv1 * tanhf(acc[ms][1][i] + bE1);
            float pt = vv0 * tanhf(acc[ms][0][i] + bT0) + vv1 * tanhf(acc[ms][1][i] + bT1);
            pe = half_reduce16(pe);
            pt = half_reduce16(pt);
            if (lm == 0) {
                atomicAdd(&sE[ms * 16 + lh * 8 + i], pe);
                atomicAdd(&sT[ms * 16 + lh * 8 + i], pt);
            }
        }
    }
    __syncthreads();

    if (tid < MT) {
        const int s = s0 + tid;
        if (s < Sc) out[(size_t)b * Sc + s] = 5.0f * sE[tid] + sT[tid];
    }
}

// ---------- launcher ----------
extern "C" void kernel_launch(void* const* d_in, const int* in_sizes, int n_in,
                              void* d_out, int out_size, void* d_ws, size_t ws_size,
                              hipStream_t stream) {
    const float* stat = (const float*)d_in[0];   // (B,S,H)
    const float* se   = (const float*)d_in[1];   // (B,1,H)
    const float* dec  = (const float*)d_in[2];   // (B,1,H)
    const float* tgt  = (const float*)d_in[3];   // (B,1,H)
    const float* lhh  = (const float*)d_in[4];   // (1,B,H)
    const float* Wih  = (const float*)d_in[5];
    const float* Whh  = (const float*)d_in[6];
    const float* bih  = (const float*)d_in[7];
    const float* bhh  = (const float*)d_in[8];
    const float* vE   = (const float*)d_in[9];
    const float* WE   = (const float*)d_in[10];  // (H, 2H)
    const float* vT   = (const float*)d_in[11];
    const float* WT   = (const float*)d_in[12];  // (H, 2H)
    const float* vP   = (const float*)d_in[13];
    const float* WP   = (const float*)d_in[14];  // (H, 3H)

    float* out = (float*)d_out;

    constexpr int BH = Bc * Hc;   // 32768
    constexpr int BS = Bc * Sc;   // 128000
    float* ws     = (float*)d_ws;
    float* hnew   = ws + 0 * BH;
    float* biasE  = ws + 1 * BH;
    float* biasT  = ws + 2 * BH;
    float* biasS  = ws + 3 * BH;
    float* biasPE = ws + 4 * BH;
    float* biasPT = ws + 5 * BH;
    float* scE    = ws + 6 * BH;
    float* scT    = ws + 6 * BH + BS;
    bf16*  WpkE   = (bf16*)(ws + 6 * BH + 2 * BS);   // 32B-aligned (offset 1,810,432 B)
    bf16*  WpkT   = WpkE + WPK_ELEMS;
    bf16*  WpkP   = WpkT + WPK_ELEMS;

    pack_w_kernel<<<16 * KT, 256, 0, stream>>>(WE, 2 * Hc, WpkE);
    pack_w_kernel<<<16 * KT, 256, 0, stream>>>(WT, 2 * Hc, WpkT);
    pack_w_kernel<<<16 * KT, 256, 0, stream>>>(WP, 3 * Hc, WpkP);
    gru_kernel<<<Bc, 256, 0, stream>>>(dec, lhh, Wih, Whh, bih, bhh, hnew, out + BS);
    bias_kernel<<<Bc, 256, 0, stream>>>(hnew, tgt, se, WE, WT, WP, biasE, biasT, biasS);
    attn_scores_kernel<<<Bc * NSTM, 256, 0, stream>>>(stat, WpkE, WpkT, biasE, biasT, vE, vT, scE, scT);
    softmax_ctx_kernel<<<Bc, 256, 0, stream>>>(stat, scE, scT, WP, biasS, biasPE, biasPT);
    ptr_kernel<<<Bc * NSTM, 256, 0, stream>>>(stat, WpkP, biasPE, biasPT, vP, out);
}
